// LatentHypergraphODE_4088808866145
// MI455X (gfx1250) — matrix operand order repaired and checked by reading, not compiled, not run on hardware
//
#include <hip/hip_runtime.h>

#define N_NODES  50000
#define N_HEDGES 10000
#define NNZ      200000
#define OBS      128
#define LAT      256
#define N_STEPS  8

typedef __attribute__((ext_vector_type(16))) __bf16 v16bf;
typedef __attribute__((ext_vector_type(8)))  float  v8f;

// pack two f32 -> two bf16 (round-to-nearest-even), b in high half
__device__ __forceinline__ unsigned pack2_bf16(float a, float b) {
  union { float f; unsigned u; } x, y;
  x.f = a; y.f = b;
  unsigned ra = (x.u + 0x7FFFu + ((x.u >> 16) & 1u)) >> 16;
  unsigned rb = (y.u + 0x7FFFu + ((y.u >> 16) & 1u)) >> 16;
  return (ra & 0xFFFFu) | (rb << 16);
}

// ---------------------------------------------------------------------------
// Tiled GEMM: C[M,N] = act( diag(rowScale) * A[M,K] @ B[K,N] + bias[N] )
// ACT: 0 = none, 1 = relu, 2 = tanh.  K % 32 == 0, N % 128 == 0.
// Block tile 128x128, 8 waves; wave tile 64x32 = 4x2 WMMA 16x16 tiles.
// LDS tiles pre-swizzled into v_wmma_f32_16x16x32_bf16 fragment layout:
//   A frag: k -> lanehalf=(k>>3)&1, e = (k&7) + 8*(k>>4), lane = r + 16*half
//   B frag: k -> lanehalf=(k>>3)&1, e = (k&7) + 8*(k>>4),           lane = n + 16*half
// ---------------------------------------------------------------------------
template<int ACT>
__global__ __launch_bounds__(256)
void gemm_bf16_wmma(const float* __restrict__ A, const float* __restrict__ B,
                    const float* __restrict__ bias, const float* __restrict__ rowScale,
                    float* __restrict__ C, int M, int K, int N) {
  __shared__ __align__(32) __bf16 sA[8][32][16];   // 8 KB
  __shared__ __align__(32) __bf16 sB[8][32][16];   // 8 KB

  const int m0   = blockIdx.x * 128;
  const int n0   = blockIdx.y * 128;
  const int tid  = threadIdx.x;
  const int lane = tid & 31;
  const int wave = tid >> 5;
  const int wm   = wave & 1;   // 0..1 : 64-row half
  const int wn   = wave >> 1;  // 0..3 : 32-col strip

  v8f acc[4][2] = {};

  for (int k0 = 0; k0 < K; k0 += 32) {
    // ---- stage A (f32 -> bf16, fold per-row scale), fragment layout -------
    // branch-free: clamp row, zero via scale when out of range
    #pragma unroll
    for (int it = tid; it < 128 * 8; it += 256) {   // 128 rows x 8 float4/row
      int row = it >> 3;
      int kq  = (it & 7) << 2;                      // 0,4,...,28
      int gr  = m0 + row;
      int cr  = gr < M ? gr : 0;
      float s = (gr < M) ? (rowScale ? rowScale[gr] : 1.0f) : 0.0f;
      float4 v = *(const float4*)(A + (size_t)cr * K + k0 + kq);
      int tile = row >> 4;
      int r    = row & 15;
      int ln   = r + (((kq >> 3) & 1) << 4);
      int e0   = (kq & 7) + ((kq >> 4) << 3);
      uint2 p;
      p.x = pack2_bf16(v.x * s, v.y * s);
      p.y = pack2_bf16(v.z * s, v.w * s);
      *(uint2*)(&sA[tile][ln][e0]) = p;             // one ds_store_b64
    }
    // ---- stage B (weights [K][N] row-major): one K-quad per thread --------
    #pragma unroll
    for (int it = tid; it < 8 * 128; it += 256) {   // 8 K-quads x 128 cols
      int kq = it >> 7;                             // 0..7  (k = kq*4..kq*4+3)
      int n  = it & 127;
      const float* Bp = B + (size_t)(k0 + kq * 4) * N + n0 + n;
      float b0 = Bp[0];
      float b1 = Bp[(size_t)N];
      float b2 = Bp[2 * (size_t)N];
      float b3 = Bp[3 * (size_t)N];
      int tile = n >> 4;
      int nn   = n & 15;
      int ln   = nn + (((kq >> 1) & 1) << 4);
      int e0   = ((kq & 1) << 2) + ((kq >> 2) << 3);
      uint2 p;
      p.x = pack2_bf16(b0, b1);
      p.y = pack2_bf16(b2, b3);
      *(uint2*)(&sB[tile][ln][e0]) = p;             // one ds_store_b64
    }
    // ---- prefetch next A K-slab into cache while we compute this one ------
    if (k0 + 32 < K) {
      int pr = m0 + (tid >> 1);
      if (pr < M)
        __builtin_prefetch(A + (size_t)pr * K + (k0 + 32) + (tid & 1) * 16, 0, 0);
    }
    __syncthreads();

    // ---- load all operand fragments, then 8 back-to-back WMMAs ------------
    v16bf afr[4], bfr[2];
    #pragma unroll
    for (int i = 0; i < 4; i++)
      afr[i] = *(const v16bf*)(&sA[wm * 4 + i][lane][0]);
    #pragma unroll
    for (int j = 0; j < 2; j++)
      bfr[j] = *(const v16bf*)(&sB[wn * 2 + j][lane][0]);
    #pragma unroll
    for (int i = 0; i < 4; i++) {
      #pragma unroll
      for (int j = 0; j < 2; j++) {
        acc[i][j] = __builtin_amdgcn_wmma_f32_16x16x32_bf16(
            false, afr[i], false, bfr[j], (short)0, acc[i][j], false, false);
      }
    }
    __syncthreads();
  }

  // ---- epilogue: bias + activation + guarded store ------------------------
  const int half   = lane >> 4;
  const int col_in = lane & 15;
  #pragma unroll
  for (int i = 0; i < 4; i++) {
    int row_base = m0 + wm * 64 + i * 16 + half * 8;
    #pragma unroll
    for (int j = 0; j < 2; j++) {
      int col  = n0 + wn * 32 + j * 16 + col_in;
      float bv = bias ? bias[col] : 0.f;
      #pragma unroll
      for (int e = 0; e < 8; e++) {
        int row = row_base + e;
        if (row < M) {
          float v = acc[i][j][e] + bv;
          if (ACT == 1) v = fmaxf(v, 0.f);
          else if (ACT == 2) v = tanhf(v);
          C[(size_t)row * N + col] = v;
        }
      }
    }
  }
}

// ---------------------------------------------------------------------------
// Support kernels
// ---------------------------------------------------------------------------
__global__ void zero_kernel(float* __restrict__ p, int n) {
  int i = blockIdx.x * blockDim.x + threadIdx.x;
  if (i < n) p[i] = 0.f;
}

__global__ void count_kernel(const int* __restrict__ idx, float* __restrict__ cnt, int nnz) {
  int i = blockIdx.x * blockDim.x + threadIdx.x;
  if (i < nnz) atomicAdd(&cnt[idx[i]], 1.0f);
}

__global__ void inv_clamp_kernel(float* __restrict__ c, int n) {
  int i = blockIdx.x * blockDim.x + threadIdx.x;
  if (i < n) c[i] = 1.0f / fmaxf(c[i], 1.0f);
}

// dst[didx[i], :] += src[sidx[i], :]; one thread = one (pair, 4 channels)
__global__ void scatter_add_kernel(const float* __restrict__ src, const int* __restrict__ sidx,
                                   const int* __restrict__ didx, float* __restrict__ dst, int nnz) {
  int t = blockIdx.x * blockDim.x + threadIdx.x;
  if (t >= nnz * 64) return;           // 64 = LAT/4
  int i = t >> 6;
  int c = (t & 63) << 2;
  int s = sidx[i];
  int d = didx[i];
  float4 v = *(const float4*)(src + (size_t)s * LAT + c);
  float* dp = dst + (size_t)d * LAT + c;
  atomicAdd(dp + 0, v.x);
  atomicAdd(dp + 1, v.y);
  atomicAdd(dp + 2, v.z);
  atomicAdd(dp + 3, v.w);
}

__global__ void rk4_stage1_kernel(const float* __restrict__ z, const float* __restrict__ k,
                                  float* __restrict__ acc, float* __restrict__ ztmp,
                                  float dt, int n) {
  int i = blockIdx.x * blockDim.x + threadIdx.x;
  if (i < n) {
    float zz = z[i], kk = k[i];
    acc[i]  = zz + (dt / 6.0f) * kk;
    ztmp[i] = zz + (0.5f * dt) * kk;
  }
}

__global__ void rk4_stage23_kernel(const float* __restrict__ z, const float* __restrict__ k,
                                   float* __restrict__ acc, float* __restrict__ ztmp,
                                   float ca, float ct, int n) {
  int i = blockIdx.x * blockDim.x + threadIdx.x;
  if (i < n) {
    float kk = k[i];
    acc[i]  += ca * kk;
    ztmp[i]  = z[i] + ct * kk;
  }
}

__global__ void rk4_final_kernel(float* __restrict__ z, const float* __restrict__ k,
                                 const float* __restrict__ acc, float cb, int n) {
  int i = blockIdx.x * blockDim.x + threadIdx.x;
  if (i < n) z[i] = acc[i] + cb * k[i];
}

// ---------------------------------------------------------------------------
extern "C" void kernel_launch(void* const* d_in, const int* in_sizes, int n_in,
                              void* d_out, int out_size, void* d_ws, size_t ws_size,
                              hipStream_t stream) {
  (void)in_sizes; (void)n_in; (void)out_size; (void)ws_size;
  const float* X      = (const float*)d_in[0];
  const float* enc_W0 = (const float*)d_in[1];
  const float* enc_b0 = (const float*)d_in[2];
  const float* enc_W1 = (const float*)d_in[3];
  const float* enc_b1 = (const float*)d_in[4];
  const float* Wne    = (const float*)d_in[5];
  const float* bne    = (const float*)d_in[6];
  const float* Wen    = (const float*)d_in[7];
  const float* ben    = (const float*)d_in[8];
  const float* dec_W0 = (const float*)d_in[9];
  const float* dec_b0 = (const float*)d_in[10];
  const float* dec_W1 = (const float*)d_in[11];
  const float* dec_b1 = (const float*)d_in[12];
  const int* node_idx = (const int*)d_in[13];
  const int* edge_idx = (const int*)d_in[14];
  float* out = (float*)d_out;

  float* ws = (float*)d_ws;
  size_t off = 0;
  float* E    = ws + off; off += (size_t)N_HEDGES * LAT;
  float* E2   = ws + off; off += (size_t)N_HEDGES * LAT;
  float* H    = ws + off; off += (size_t)N_NODES * LAT;   // hidden / node-aggregate buffer
  float* z    = ws + off; off += (size_t)N_NODES * LAT;
  float* acc  = ws + off; off += (size_t)N_NODES * LAT;
  float* ztmp = ws + off; off += (size_t)N_NODES * LAT;
  float* kbuf = ws + off; off += (size_t)N_NODES * LAT;
  float* einv = ws + off; off += N_HEDGES;
  float* ninv = ws + off; off += N_NODES;

  auto zero = [&](float* p, int n) {
    zero_kernel<<<(n + 255) / 256, 256, 0, stream>>>(p, n);
  };
  auto gemm = [&](const float* A, const float* B, const float* bias, const float* rs,
                  float* C, int M, int K, int N, int act) {
    dim3 grid((M + 127) / 128, N / 128);
    if (act == 0)      gemm_bf16_wmma<0><<<grid, 256, 0, stream>>>(A, B, bias, rs, C, M, K, N);
    else if (act == 1) gemm_bf16_wmma<1><<<grid, 256, 0, stream>>>(A, B, bias, rs, C, M, K, N);
    else               gemm_bf16_wmma<2><<<grid, 256, 0, stream>>>(A, B, bias, rs, C, M, K, N);
  };

  // ---- incidence counts -> reciprocals (recomputed each call: deterministic) ----
  zero(einv, N_HEDGES);
  zero(ninv, N_NODES);
  count_kernel<<<(NNZ + 255) / 256, 256, 0, stream>>>(edge_idx, einv, NNZ);
  count_kernel<<<(NNZ + 255) / 256, 256, 0, stream>>>(node_idx, ninv, NNZ);
  inv_clamp_kernel<<<(N_HEDGES + 255) / 256, 256, 0, stream>>>(einv, N_HEDGES);
  inv_clamp_kernel<<<(N_NODES + 255) / 256, 256, 0, stream>>>(ninv, N_NODES);

  // ---- encoder: z0 = relu(X @ W0 + b0) @ W1 + b1 ----
  gemm(X, enc_W0, enc_b0, nullptr, H, N_NODES, OBS, LAT, 1);
  gemm(H, enc_W1, enc_b1, nullptr, z, N_NODES, LAT, LAT, 0);

  const int scat_blocks = (NNZ * 64 + 255) / 256;
  auto feval = [&](const float* zin, float* kout) {
    // node -> hyperedge (mean folded into GEMM row scale)
    zero(E, N_HEDGES * LAT);
    scatter_add_kernel<<<scat_blocks, 256, 0, stream>>>(zin, node_idx, edge_idx, E, NNZ);
    gemm(E, Wne, bne, einv, E2, N_HEDGES, LAT, LAT, 0);
    // hyperedge -> node, then tanh fused in epilogue
    zero(H, N_NODES * LAT);
    scatter_add_kernel<<<scat_blocks, 256, 0, stream>>>(E2, edge_idx, node_idx, H, NNZ);
    gemm(H, Wen, ben, ninv, kout, N_NODES, LAT, LAT, 2);
  };

  // ---- RK4 over [0,1], 8 steps ----
  const float dt = 1.0f / (float)N_STEPS;
  const int   n  = N_NODES * LAT;
  const int   nb = (n + 255) / 256;
  for (int s = 0; s < N_STEPS; s++) {
    feval(z, kbuf);
    rk4_stage1_kernel<<<nb, 256, 0, stream>>>(z, kbuf, acc, ztmp, dt, n);
    feval(ztmp, kbuf);
    rk4_stage23_kernel<<<nb, 256, 0, stream>>>(z, kbuf, acc, ztmp, dt / 3.0f, dt * 0.5f, n);
    feval(ztmp, kbuf);
    rk4_stage23_kernel<<<nb, 256, 0, stream>>>(z, kbuf, acc, ztmp, dt / 3.0f, dt, n);
    feval(ztmp, kbuf);
    rk4_final_kernel<<<nb, 256, 0, stream>>>(z, kbuf, acc, dt / 6.0f, n);
  }

  // ---- decoder: out = relu(z1 @ dW0 + db0) @ dW1 + db1 ----
  gemm(z, dec_W0, dec_b0, nullptr, H, N_NODES, LAT, LAT, 1);
  gemm(H, dec_W1, dec_b1, nullptr, out, N_NODES, LAT, OBS, 0);
}
